// ResidualBlock4D_21431886807867
// MI455X (gfx1250) — matrix operand order repaired and checked by reading, not compiled, not run on hardware
//
#include <hip/hip_runtime.h>
#include <hip/hip_bf16.h>

typedef unsigned int u32;
typedef __attribute__((ext_vector_type(4)))  u32    v4u;
typedef __attribute__((ext_vector_type(8)))  u32    v8u;
typedef __attribute__((ext_vector_type(4)))  int    v4i;
typedef __attribute__((ext_vector_type(8)))  int    v8i;
typedef __attribute__((ext_vector_type(16))) __bf16 v16bf;
typedef __attribute__((ext_vector_type(8)))  float  v8f;

#define NC       64
#define SPATIAL  65536            // 16^4
#define PD       18               // padded dim
#define PSPAT    (18*18*18*18)    // 104976
#define KTAPS    81
#define NGROUP   32
#define GELEMS   (2*SPATIAL)      // channels-per-group * spatial = 131072

#define PLANE_E  (PD*PD*NC)       // 18*18*64 = 20736 elements, contiguous in hpad
#define ROWS     (PD*PD)          // 324 rows of 64 elements (128 B) each
#define RSTRIDE  72               // LDS row stride in elements (128B data + 16B pad)

#if defined(__has_builtin)
#if __has_builtin(__builtin_amdgcn_tensor_load_to_lds) && __has_builtin(__builtin_amdgcn_s_wait_tensorcnt)
#define USE_TDM 1
#endif
#endif

// ---------------- weight repack: w[co][ci][t] f32 -> wb[t][co][ci] bf16 ----------------
__global__ void repack_w(const float* __restrict__ w, __bf16* __restrict__ wb) {
    int idx = blockIdx.x * 256 + threadIdx.x;           // 81*64*64 = 331776 exact
    int ci = idx & 63, co = (idx >> 6) & 63, t = idx >> 12;
    wb[idx] = (__bf16)w[(co * NC + ci) * KTAPS + t];
}

// ---------------- per-group mean / rstd over [c][spatial] layout ----------------
__global__ void gn_stats(const float* __restrict__ src, float* __restrict__ stats) {
    int g = blockIdx.x, tid = threadIdx.x;
    const float* p = src + (size_t)g * GELEMS;          // group channels are contiguous
    float s = 0.f, sq = 0.f;
    for (int i = tid; i < GELEMS; i += 256) { float v = p[i]; s += v; sq += v * v; }
    __shared__ float rs[256], rq[256];
    rs[tid] = s; rq[tid] = sq; __syncthreads();
    for (int o = 128; o > 0; o >>= 1) {
        if (tid < o) { rs[tid] += rs[tid + o]; rq[tid] += rq[tid + o]; }
        __syncthreads();
    }
    if (tid == 0) {
        float m = rs[0] * (1.f / GELEMS);
        float var = rq[0] * (1.f / GELEMS) - m * m;
        stats[2 * g] = m;
        stats[2 * g + 1] = rsqrtf(var + 1e-6f);
    }
}

// ------- GN + swish + bf16 pack into zero-haloed channels-last buffer [18^4][64] -------
__global__ void gn_swish_pack(const float* __restrict__ src, const float* __restrict__ stats,
                              const float* __restrict__ gamma, const float* __restrict__ beta,
                              __bf16* __restrict__ hpad) {
    int idx = blockIdx.x * 256 + threadIdx.x;           // PSPAT*64 = 6718464 exact
    int c = idx & 63;
    int pos = idx >> 6;
    int pd4 = pos % PD; pos /= PD;
    int pd3 = pos % PD; pos /= PD;
    int pd2 = pos % PD;
    int pd1 = pos / PD;
    float o = 0.f;
    if (pd1 >= 1 && pd1 <= 16 && pd2 >= 1 && pd2 <= 16 &&
        pd3 >= 1 && pd3 <= 16 && pd4 >= 1 && pd4 <= 16) {
        int p = (((pd1 - 1) * 16 + (pd2 - 1)) * 16 + (pd3 - 1)) * 16 + (pd4 - 1);
        float v = src[c * SPATIAL + p];
        int g = c >> 1;
        float vn = (v - stats[2 * g]) * stats[2 * g + 1] * gamma[c] + beta[c];
        o = vn / (1.f + __expf(-vn));                   // swish
    }
    hpad[idx] = (__bf16)o;
}

// ---------------- fragment loaders matched to ISA lane layouts ----------------
__device__ __forceinline__ v16bf loadA(const __bf16* p) {
    // lane needs K chunk [0..7] and [16..23] (lane-half adds +8): two contiguous 16B loads
    const v4u* q = (const v4u*)p;
    v4u lo = q[0], hi = q[2];                           // +32 bytes
    v8u u; u[0] = lo[0]; u[1] = lo[1]; u[2] = lo[2]; u[3] = lo[3];
           u[4] = hi[0]; u[5] = hi[1]; u[6] = hi[2]; u[7] = hi[3];
    return __builtin_bit_cast(v16bf, u);
}
__device__ __forceinline__ v16bf loadB(const __bf16* p) {
    // lane needs K chunk [0..7] and [16..23] (lane-half adds +8) of one column: two 16B loads
    const v4u* q = (const v4u*)p;
    v4u lo = q[0], hi = q[2];
    v8u u; u[0] = lo[0]; u[1] = lo[1]; u[2] = lo[2]; u[3] = lo[3];
           u[4] = hi[0]; u[5] = hi[1]; u[6] = hi[2]; u[7] = hi[3];
    return __builtin_bit_cast(v16bf, u);
}
__device__ __forceinline__ v8f wmma_bf16(v16bf a, v16bf b, v8f c) {
    return __builtin_amdgcn_wmma_f32_16x16x32_bf16(false, a, false, b, (short)0, c, false, false);
}

#ifdef USE_TDM
// 1-D TDM copy: PLANE_E bf16 elements, padded into LDS (16B pad every 128B -> 144B rows).
// D# built per ISA ch.8: group0 = {count, lds_addr, global_addr[56:0], type=2},
// group1 = {data_size=2B, pad cfg, tensor_dim0 = tile_dim0 = nelem}.
// Toolchain exposes the 6-arg builtin: (v4u g0, v8i g1, v4i g2, v4i g3, v8i extra, i32 cpol);
// groups 2/3 and the trailing group are zero for a <=2-D tensor.
__device__ __forceinline__ void tdm_load_plane(u32 lds_off, const __bf16* gsrc) {
    unsigned long long ga = (unsigned long long)(uintptr_t)gsrc;
    const u32 n = PLANE_E;
    v4u g0;
    g0[0] = 1u;                                          // count=1, user descriptor
    g0[1] = lds_off;                                     // lds_addr (bytes)
    g0[2] = (u32)ga;                                     // global_addr[31:0]
    g0[3] = (u32)((ga >> 32) & 0x01FFFFFFu) | 0x80000000u; // ga[56:32] | type=2
    v8i g1;
    g1[0] = (int)((1u << 16)        // data_size = 1 -> 2 bytes
                | (1u << 20)        // pad_enable
                | (4u << 22)        // pad_interval: every 32 DWORDs (128 B)
                | (3u << 25));      // pad_amount: 4 DWORDs (16 B)
    g1[1] = (int)((n & 0xFFFFu) << 16);                  // tensor_dim0[15:0]
    g1[2] = (int)((n >> 16) | (1u << 16));               // tensor_dim0[31:16] | tensor_dim1=1
    g1[3] = (int)((n & 0xFFFFu) << 16);                  // tile_dim0 = n (1-D tile)
    g1[4] = 0;                                           // tile_dim1/2 unused
    g1[5] = (int)n;                                      // tensor_dim0_stride (unused for 1 line)
    g1[6] = 0;
    g1[7] = 0;
    v4i z4 = {0, 0, 0, 0};
    v8i z8 = {0, 0, 0, 0, 0, 0, 0, 0};
    __builtin_amdgcn_tensor_load_to_lds(g0, g1, z4, z4, z8, 0);
}
#endif

// -------- implicit-GEMM Conv4D, one (d1,d2) plane per WG, A staged in LDS ----------
// out[co][p] = sum_{tap,ci} hpad * wb  (+bias, optional +residual)
template <bool RES>
__global__ void conv_wmma(const __bf16* __restrict__ hp, const __bf16* __restrict__ wb,
                          const float* __restrict__ bias, const float* __restrict__ res,
                          float* __restrict__ out) {
    __shared__ __bf16 sA[ROWS * RSTRIDE];               // 324*72*2 = 46656 B, padded rows

    const int lane  = threadIdx.x & 31;
    const int wave  = threadIdx.x >> 5;                 // 8 waves: d3 pairs 0..15
    const int Mbase = blockIdx.x * 256 + wave * 32;
    const int m  = lane & 15;
    const int hh = lane >> 4;
    const int d2 = blockIdx.x & 15;
    const int d1 = blockIdx.x >> 4;
    const int w2 = wave * 2;                            // this wave's first d3 row

    v8f acc[2][4] = {};                                 // 2 M-frags x 4 N-frags (N=64)

    const int bLane = m * 64 + hh * 8;                  // B lane offset (elements)

#pragma unroll 1
    for (int s = 0; s < 9; ++s) {                       // (t1,t2) stages
        const int t1 = s / 3, t2 = s % 3;
        const __bf16* gsrc = hp + ((d1 + t1) * PD + (d2 + t2)) * PLANE_E;

        __syncthreads();                                // previous stage fully consumed
#ifdef USE_TDM
        if (wave == 0) {                                // TDM is a per-wave DMA op
            tdm_load_plane((u32)(uintptr_t)(void*)sA, gsrc);
            __builtin_amdgcn_s_wait_tensorcnt((short)0);
        }
#else
        for (int i = threadIdx.x; i < ROWS * 8; i += 256) {   // 324 rows x 8 x 16B
            int row = i >> 3, part = i & 7;
            *(v4u*)(sA + row * RSTRIDE + part * 8) =
                *(const v4u*)(gsrc + row * NC + part * 8);
        }
#endif
        __syncthreads();                                // tile visible to all waves

#pragma unroll 1
        for (int t3 = 0; t3 < 3; ++t3) {
#pragma unroll
            for (int t4 = 0; t4 < 3; ++t4) {
                const int tap = s * 9 + t3 * 3 + t4;
                const __bf16* wt = wb + tap * 4096;
                // LDS rows: pd3 = d3f + t3, pd4 = m + t4  (conflict-free: 144B stride)
                const int a0 = ((w2     + t3) * PD + m + t4) * RSTRIDE + hh * 8;
                const int a1 = ((w2 + 1 + t3) * PD + m + t4) * RSTRIDE + hh * 8;
#pragma unroll
                for (int ch = 0; ch < 2; ++ch) {        // K = 64 ci per tap, 2 x K32
                    const int ko = ch * 32;
                    v16bf A0 = loadA(sA + a0 + ko);     // ds_load_b128 x2
                    v16bf A1 = loadA(sA + a1 + ko);
                    v16bf B0 = loadB(wt + bLane + ko);  // global, WGP$-resident
                    v16bf B1 = loadB(wt + bLane + 1024 + ko);
                    v16bf B2 = loadB(wt + bLane + 2048 + ko);
                    v16bf B3 = loadB(wt + bLane + 3072 + ko);
                    acc[0][0] = wmma_bf16(A0, B0, acc[0][0]);
                    acc[0][1] = wmma_bf16(A0, B1, acc[0][1]);
                    acc[0][2] = wmma_bf16(A0, B2, acc[0][2]);
                    acc[0][3] = wmma_bf16(A0, B3, acc[0][3]);
                    acc[1][0] = wmma_bf16(A1, B0, acc[1][0]);
                    acc[1][1] = wmma_bf16(A1, B1, acc[1][1]);
                    acc[1][2] = wmma_bf16(A1, B2, acc[1][2]);
                    acc[1][3] = wmma_bf16(A1, B3, acc[1][3]);
                }
            }
        }
    }

    // epilogue: C/D layout -> lane holds M = r(+8 for hi half), fixed N; contiguous stores
#pragma unroll
    for (int f = 0; f < 2; ++f)
#pragma unroll
    for (int nf = 0; nf < 4; ++nf) {
        const int co = nf * 16 + m;
        const int pb = Mbase + f * 16 + hh * 8;
        const float bv = bias[co];
        float* o = out + co * SPATIAL + pb;
#pragma unroll
        for (int q = 0; q < 8; ++q) {
            float v = acc[f][nf][q] + bv;
            if constexpr (RES) v += res[co * SPATIAL + pb + q];
            o[q] = v;
        }
    }
}

extern "C" void kernel_launch(void* const* d_in, const int* in_sizes, int n_in,
                              void* d_out, int out_size, void* d_ws, size_t ws_size,
                              hipStream_t stream) {
    const float* x     = (const float*)d_in[0];
    const float* gn1_g = (const float*)d_in[1];
    const float* gn1_b = (const float*)d_in[2];
    const float* w1    = (const float*)d_in[3];
    const float* b1    = (const float*)d_in[4];
    const float* gn2_g = (const float*)d_in[5];
    const float* gn2_b = (const float*)d_in[6];
    const float* w2    = (const float*)d_in[7];
    const float* b2    = (const float*)d_in[8];
    float* out = (float*)d_out;

    char* ws = (char*)d_ws;                       // all offsets 128B-aligned
    __bf16* wb1    = (__bf16*)(ws);                                 //   663,552 B
    __bf16* wb2    = (__bf16*)(ws + 663552);                        //   663,552 B
    float*  stats1 = (float*)(ws + 1327104);                        //       256 B
    float*  stats2 = (float*)(ws + 1327360);                        //       256 B
    __bf16* hpad   = (__bf16*)(ws + 1327616);                       // 13,436,928 B (reused)
    float*  y1     = (float*)(ws + 1327616 + 13436928);             // 16,777,216 B

    // repack both conv weights (81*64*64 / 256 = 1296 blocks, exact)
    repack_w<<<1296, 256, 0, stream>>>(w1, wb1);
    repack_w<<<1296, 256, 0, stream>>>(w2, wb2);

    // GN1 -> swish -> bf16 padded pack
    gn_stats<<<NGROUP, 256, 0, stream>>>(x, stats1);
    gn_swish_pack<<<PSPAT * NC / 256, 256, 0, stream>>>(x, stats1, gn1_g, gn1_b, hpad);

    // conv1 (+b1) -> y1 (fp32, channel-major)
    conv_wmma<false><<<SPATIAL / 256, 256, 0, stream>>>(hpad, wb1, b1, nullptr, y1);

    // GN2 -> swish -> bf16 padded pack (hpad reused; conv1 already complete on-stream)
    gn_stats<<<NGROUP, 256, 0, stream>>>(y1, stats2);
    gn_swish_pack<<<PSPAT * NC / 256, 256, 0, stream>>>(y1, stats2, gn2_g, gn2_b, hpad);

    // conv2 (+b2) + residual x -> out
    conv_wmma<true><<<SPATIAL / 256, 256, 0, stream>>>(hpad, wb2, b2, x, out);
}
